// RNNCRF_56839597195864
// MI455X (gfx1250) — matrix operand order repaired and checked by reading, not compiled, not run on hardware
//
#include <hip/hip_runtime.h>
#include <math.h>

// ---------------- problem constants ----------------
#define SS 256      // sequence length
#define BB 32       // batch
#define EE 256      // embedding dim
#define HH 512      // hidden dim
#define G4H 2048    // 4*H
#define KCAT 768    // E + H (fused input+recurrent K)
#define TT 10       // tags
#define TPAD 16     // tags padded to one WMMA N tile
#define START_TAG 8
#define STOP_TAG 9
#define NEGV (-10000.0f)

// ---------------- WMMA types ----------------
typedef __attribute__((ext_vector_type(16))) __bf16 bf16x16;
typedef __attribute__((ext_vector_type(8)))  __bf16 bf16x8;
typedef __attribute__((ext_vector_type(8)))  float  f32x8;

__device__ __forceinline__ __bf16 f2bf(float f) {
    union { float f; unsigned u; } v; v.f = f;
    unsigned r = (v.u + 0x7FFFu + ((v.u >> 16) & 1u)) >> 16;
    union { unsigned short s; __bf16 b; } o; o.s = (unsigned short)r;
    return o.b;
}

// A fragment: 16x32 bf16, row-major source, rows m0..m0+15, cols k0..k0+31.
__device__ __forceinline__ bf16x16 load_afrag(const __bf16* base, int ld, int m0, int k0, int lane) {
    int m = m0 + (lane & 15);
    int h = lane >> 4;
    const __bf16* p = base + (size_t)m * ld + k0 + h * 8;
    bf16x8 lo = *(const bf16x8*)(p);
    bf16x8 hi = *(const bf16x8*)(p + 16);
    return __builtin_shufflevector(lo, hi, 0,1,2,3,4,5,6,7,8,9,10,11,12,13,14,15);
}

// B fragment: 32x16 bf16 (K x N). Source stored [N][K] row-major.
__device__ __forceinline__ bf16x16 load_bfrag(const __bf16* base, int ld, int n0, int k0, int lane) {
    int n = n0 + (lane & 15);
    int h = lane >> 4;
    const __bf16* p = base + (size_t)n * ld + k0 + h * 8;
    bf16x8 lo = *(const bf16x8*)(p);
    bf16x8 hi = *(const bf16x8*)(p + 16);
    return __builtin_shufflevector(lo, hi, 0,1,2,3,4,5,6,7,8,9,10,11,12,13,14,15);
}

__device__ __forceinline__ f32x8 wmma_bf16(bf16x16 a, bf16x16 b, f32x8 c) {
    return __builtin_amdgcn_wmma_f32_16x16x32_bf16(false, a, false, b, (short)0, c, false, false);
}

__device__ __forceinline__ float sigmoidf_(float x) { return 1.0f / (1.0f + expf(-x)); }

// ---------------- kernel 1: gather embeddings + build bf16 weight buffers ----------------
__global__ void prep_kernel(const int* __restrict__ tokens, const float* __restrict__ emb,
                            const float* __restrict__ wih_f, const float* __restrict__ whh_f,
                            const float* __restrict__ wih_b, const float* __restrict__ whh_b,
                            const float* __restrict__ w_out,
                            __bf16* __restrict__ x_bf, __bf16* __restrict__ wcat,
                            __bf16* __restrict__ woutp) {
    const size_t NX = (size_t)SS * BB * EE;            // 2,097,152
    const size_t NW = (size_t)2 * G4H * KCAT;          // 3,145,728
    const size_t NO = (size_t)TPAD * (2 * HH);         // 16,384
    const size_t total = NX + NW + NO;
    size_t stride = (size_t)gridDim.x * blockDim.x;
    for (size_t idx = (size_t)blockIdx.x * blockDim.x + threadIdx.x; idx < total; idx += stride) {
        if (idx < NX) {
            size_t e  = idx & (EE - 1);
            size_t sb = idx >> 8;          // s*32 + b
            int b = (int)(sb & 31);
            int s = (int)(sb >> 5);
            int tok = tokens[(size_t)b * SS + s];
            x_bf[idx] = f2bf(emb[(size_t)tok * EE + e]);
        } else if (idx < NX + NW) {
            size_t j = idx - NX;
            int d = (j >= (size_t)G4H * KCAT);
            size_t jj = d ? j - (size_t)G4H * KCAT : j;
            int g = (int)(jj / KCAT);
            int k = (int)(jj % KCAT);
            const float* wi = d ? wih_b : wih_f;
            const float* wh = d ? whh_b : whh_f;
            float v = (k < EE) ? wi[(size_t)g * EE + k] : wh[(size_t)g * HH + (k - EE)];
            wcat[j] = f2bf(v);
        } else {
            size_t j = idx - NX - NW;
            int t = (int)(j / (2 * HH));
            woutp[j] = f2bf(t < TT ? w_out[j] : 0.0f);
        }
    }
}

// ---------------- kernel 2: persistent BiLSTM, one workgroup per direction ----------------
// grid=2 (dir), block=512 (16 waves -> ~256 VGPRs/wave, no spills).
// Each wave owns TWO 16-wide column slices (jslice = wave and wave+16), processed as two
// passes per step with 8 accumulators live per pass. h lives in LDS (bf16); c in registers.
// New h is parked in registers and committed to LDS only after all waves finished reading.
__global__ void __launch_bounds__(512)
lstm_kernel(const __bf16* __restrict__ x_bf, const __bf16* __restrict__ wcat,
            const float* __restrict__ bias_f, const float* __restrict__ bias_b,
            __bf16* __restrict__ hout_f, __bf16* __restrict__ hout_b) {
    const int dir = blockIdx.x;
    const __bf16* W = wcat + (size_t)dir * G4H * KCAT;
    const float* bias = dir ? bias_b : bias_f;
    __bf16* hout = dir ? hout_b : hout_f;

    __shared__ __bf16 h_lds[BB][HH];   // 32 KB

    const int tid  = threadIdx.x;
    const int wave = tid >> 5;         // 0..15
    const int lane = tid & 31;
    const int half = lane >> 4;
    const int lq   = lane & 15;

    // zero initial hidden state
    for (int i = tid; i < BB * HH; i += 512) ((__bf16*)h_lds)[i] = f2bf(0.0f);
    __syncthreads();

    // per-thread cell state: creg[pass][mt][r] for batch row b = mt*16+r+8*half,
    // column j = (wave + pass*16)*16 + lq
    float creg[2][2][8];
    float bi[2], bff[2], bg[2], bo[2];
#pragma unroll
    for (int p = 0; p < 2; ++p) {
        const int jp = (wave + p * 16) * 16 + lq;
        bi[p]  = bias[jp];
        bff[p] = bias[HH + jp];
        bg[p]  = bias[2 * HH + jp];
        bo[p]  = bias[3 * HH + jp];
#pragma unroll
        for (int mt = 0; mt < 2; ++mt)
#pragma unroll
            for (int r = 0; r < 8; ++r) creg[p][mt][r] = 0.0f;
    }

    for (int step = 0; step < SS; ++step) {
        const int t = dir ? (SS - 1 - step) : step;
        const __bf16* xrows = x_bf + (size_t)t * BB * EE;  // rows t*32 .. t*32+31, ld=EE
        __bf16 hsave[2][2][8];

#pragma unroll
        for (int p = 0; p < 2; ++p) {
            const int js = wave + p * 16;          // 0..31 across passes
            f32x8 acc[2][4] = {};                  // [m-tile][gate]

            // x part: K = 0..255 from global
#pragma unroll 2
            for (int kk = 0; kk < EE; kk += 32) {
                bf16x16 a0 = load_afrag(xrows, EE, 0,  kk, lane);
                bf16x16 a1 = load_afrag(xrows, EE, 16, kk, lane);
#pragma unroll
                for (int g = 0; g < 4; ++g) {
                    bf16x16 bfr = load_bfrag(W, KCAT, g * HH + js * 16, kk, lane);
                    acc[0][g] = wmma_bf16(a0, bfr, acc[0][g]);
                    acc[1][g] = wmma_bf16(a1, bfr, acc[1][g]);
                }
            }
            // h part: K = 256..767, A from LDS
#pragma unroll 2
            for (int kk = 0; kk < HH; kk += 32) {
                bf16x16 a0 = load_afrag(&h_lds[0][0], HH, 0,  kk, lane);
                bf16x16 a1 = load_afrag(&h_lds[0][0], HH, 16, kk, lane);
#pragma unroll
                for (int g = 0; g < 4; ++g) {
                    bf16x16 bfr = load_bfrag(W, KCAT, g * HH + js * 16, EE + kk, lane);
                    acc[0][g] = wmma_bf16(a0, bfr, acc[0][g]);
                    acc[1][g] = wmma_bf16(a1, bfr, acc[1][g]);
                }
            }

            const int j = js * 16 + lq;
#pragma unroll
            for (int mt = 0; mt < 2; ++mt) {
#pragma unroll
                for (int r = 0; r < 8; ++r) {
                    const int b = mt * 16 + r + 8 * half;   // batch row
                    float ig = acc[mt][0][r] + bi[p];
                    float fg = acc[mt][1][r] + bff[p];
                    float gg = acc[mt][2][r] + bg[p];
                    float og = acc[mt][3][r] + bo[p];
                    float c = sigmoidf_(fg) * creg[p][mt][r] + sigmoidf_(ig) * tanhf(gg);
                    creg[p][mt][r] = c;
                    float h = sigmoidf_(og) * tanhf(c);
                    __bf16 hb = f2bf(h);
                    hsave[p][mt][r] = hb;
                    hout[((size_t)t * BB + b) * HH + j] = hb;  // global, not read this phase
                }
            }
        }

        __syncthreads();   // all waves finished reading old h_lds
#pragma unroll
        for (int p = 0; p < 2; ++p) {
            const int j = (wave + p * 16) * 16 + lq;
#pragma unroll
            for (int mt = 0; mt < 2; ++mt)
#pragma unroll
                for (int r = 0; r < 8; ++r)
                    h_lds[mt * 16 + r + 8 * half][j] = hsave[p][mt][r];
        }
        __syncthreads();   // new h visible before next step's reads
    }
}

// ---------------- kernel 3: output projection (WMMA, N padded to 16) ----------------
__global__ void __launch_bounds__(256)
proj_kernel(const __bf16* __restrict__ hf, const __bf16* __restrict__ hb,
            const __bf16* __restrict__ woutp, const float* __restrict__ b_out,
            float* __restrict__ feats) {
    const int wave = blockIdx.x * 8 + (threadIdx.x >> 5);  // 0..511 (m-tiles)
    const int lane = threadIdx.x & 31;
    const int m0 = wave * 16;
    f32x8 acc = {};
    for (int kk = 0; kk < 2 * HH; kk += 32) {
        const __bf16* abase = (kk < HH) ? hf : hb;
        int k = (kk < HH) ? kk : kk - HH;
        bf16x16 a = load_afrag(abase, HH, m0, k, lane);
        bf16x16 b = load_bfrag(woutp, 2 * HH, 0, kk, lane);
        acc = wmma_bf16(a, b, acc);
    }
    const int n = lane & 15;
    const int half = lane >> 4;
    float bo = (n < TT) ? b_out[n] : 0.0f;
#pragma unroll
    for (int r = 0; r < 8; ++r) {
        int m = m0 + r + 8 * half;
        feats[(size_t)m * TPAD + n] = acc[r] + bo;
    }
}

// ---------------- kernel 4: CRF forward (single block, thread = (batch, tag)) ----------------
__global__ void __launch_bounds__(320)
crf_kernel(const float* __restrict__ feats, const float* __restrict__ trans,
           const int* __restrict__ lengths, float* __restrict__ out) {
    __shared__ float fv[BB][TT];
    __shared__ float tr[TT][TT];
    __shared__ float term[BB][TT];
    const int tid = threadIdx.x;       // 0..319
    const int b   = tid & 31;
    const int tag = tid >> 5;          // 0..9
    if (tid < TT * TT) tr[tid / TT][tid % TT] = trans[tid];
    fv[b][tag] = (tag == START_TAG) ? 0.0f : NEGV;
    const int len = lengths[b];
    __syncthreads();

    for (int s = 0; s < SS; ++s) {
        float v[TT];
        float mx = -3.4e38f;
#pragma unroll
        for (int p = 0; p < TT; ++p) { v[p] = fv[b][p] + tr[tag][p]; mx = fmaxf(mx, v[p]); }
        float sum = 0.0f;
#pragma unroll
        for (int p = 0; p < TT; ++p) sum += expf(v[p] - mx);
        float nv = mx + logf(sum) + feats[((size_t)s * BB + b) * TPAD + tag];
        __syncthreads();
        if (s < len) fv[b][tag] = nv;
        __syncthreads();
    }

    term[b][tag] = fv[b][tag] + tr[STOP_TAG][tag];
    __syncthreads();
    if (tid < BB) {
        float mx = -3.4e38f;
#pragma unroll
        for (int p = 0; p < TT; ++p) mx = fmaxf(mx, term[tid][p]);
        float sum = 0.0f;
#pragma unroll
        for (int p = 0; p < TT; ++p) sum += expf(term[tid][p] - mx);
        out[tid] = (mx + logf(sum)) / (float)lengths[tid];
    }
}

// ---------------- workspace layout (bytes) ----------------
#define OFF_X   ((size_t)0)                         // x_bf   : S*B*E bf16      = 4,194,304
#define OFF_W   (OFF_X  + (size_t)SS*BB*EE*2)       // wcat   : 2*4H*KCAT bf16  = 6,291,456
#define OFF_WO  (OFF_W  + (size_t)2*G4H*KCAT*2)     // woutp  : 16*1024 bf16    = 32,768
#define OFF_HF  (OFF_WO + (size_t)TPAD*2*HH*2)      // hf     : S*B*H bf16      = 8,388,608
#define OFF_HB  (OFF_HF + (size_t)SS*BB*HH*2)       // hb     : S*B*H bf16      = 8,388,608
#define OFF_FE  (OFF_HB + (size_t)SS*BB*HH*2)       // feats  : S*B*16 f32      = 524,288
// total ~27.8 MB

extern "C" void kernel_launch(void* const* d_in, const int* in_sizes, int n_in,
                              void* d_out, int out_size, void* d_ws, size_t ws_size,
                              hipStream_t stream) {
    (void)in_sizes; (void)n_in; (void)out_size; (void)ws_size;
    const int*   tokens = (const int*)d_in[0];
    const int*   lengths= (const int*)d_in[1];
    const float* emb    = (const float*)d_in[2];
    const float* wih_f  = (const float*)d_in[3];
    const float* whh_f  = (const float*)d_in[4];
    const float* b_f    = (const float*)d_in[5];
    const float* wih_b  = (const float*)d_in[6];
    const float* whh_b  = (const float*)d_in[7];
    const float* b_b    = (const float*)d_in[8];
    const float* w_out  = (const float*)d_in[9];
    const float* b_out  = (const float*)d_in[10];
    const float* trans  = (const float*)d_in[11];
    float* out = (float*)d_out;

    char* ws = (char*)d_ws;
    __bf16* x_bf  = (__bf16*)(ws + OFF_X);
    __bf16* wcat  = (__bf16*)(ws + OFF_W);
    __bf16* woutp = (__bf16*)(ws + OFF_WO);
    __bf16* hf    = (__bf16*)(ws + OFF_HF);
    __bf16* hb    = (__bf16*)(ws + OFF_HB);
    float*  feats = (float*)(ws + OFF_FE);

    prep_kernel<<<2048, 256, 0, stream>>>(tokens, emb, wih_f, whh_f, wih_b, whh_b,
                                          w_out, x_bf, wcat, woutp);
    lstm_kernel<<<2, 512, 0, stream>>>(x_bf, wcat, b_f, b_b, hf, hb);
    proj_kernel<<<64, 256, 0, stream>>>(hf, hb, woutp, b_out, feats);
    crf_kernel<<<1, 320, 0, stream>>>(feats, trans, lengths, out);
}
